// GATv2_59407987638621
// MI455X (gfx1250) — matrix operand order repaired and checked by reading, not compiled, not run on hardware
//
#include <hip/hip_runtime.h>
#include <math.h>

// ---------------------------------------------------------------------------
// GATv2 (3 layers) for MI455X / gfx1250 (wave32, WMMA)
//   - GEMMs via v_wmma_f32_16x16x32_f16 (f16 inputs, f32 accumulate)
//   - edge phase: L2-resident gather + f32 atomics (XL/XR/acc all fit in 192MB L2)
// ---------------------------------------------------------------------------

static constexpr int N_NODES = 50000;   // divisible by 16 (3125 tiles)
static constexpr int H = 4;

typedef __attribute__((ext_vector_type(16))) _Float16 v16h;
typedef __attribute__((ext_vector_type(8)))  _Float16 v8h;
typedef __attribute__((ext_vector_type(8)))  float    v8f;

// Order-preserving float<->uint mapping for atomicMax on floats.
__device__ __forceinline__ unsigned int encode_f32(float f) {
    unsigned int u = __float_as_uint(f);
    return (u & 0x80000000u) ? ~u : (u | 0x80000000u);
}
__device__ __forceinline__ float decode_f32(unsigned int e) {
    unsigned int u = (e & 0x80000000u) ? (e & 0x7FFFFFFFu) : ~e;
    return __uint_as_float(u);
}
// encode(-inf) == 0x007FFFFF
static constexpr unsigned int ENC_NEG_INF = 0x007FFFFFu;

// ---------------------------- utility kernels ------------------------------

__global__ void fill_u32_kernel(unsigned int* __restrict__ p, unsigned int v, int n) {
    int i = blockIdx.x * blockDim.x + threadIdx.x;
    if (i < n) p[i] = v;
}

__global__ void build_edges_kernel(const int* __restrict__ ei, int* __restrict__ src,
                                   int* __restrict__ dst, int E, int n) {
    int i = blockIdx.x * blockDim.x + threadIdx.x;
    if (i < E)            { src[i] = ei[i];     dst[i] = ei[E + i]; }
    else if (i < E + n)   { src[i] = i - E;     dst[i] = i - E;     }   // self loops
}

__global__ void cvt_f16_kernel(const float* __restrict__ in, _Float16* __restrict__ out, int n) {
    int i = blockIdx.x * blockDim.x + threadIdx.x;
    if (i < n) out[i] = (_Float16)in[i];
}

// W [K, Nout] f32 (row-major)  ->  WT [Nout, K] f16 (row-major, i.e. B transposed)
__global__ void cvt_w_trans_kernel(const float* __restrict__ W, _Float16* __restrict__ WT,
                                   int K, int Nout) {
    int i = blockIdx.x * blockDim.x + threadIdx.x;
    if (i >= K * Nout) return;
    int k = i / Nout, c = i % Nout;
    WT[c * K + k] = (_Float16)W[i];
}

// ------------------------------ WMMA GEMM ----------------------------------
// C[M,Nn] = A[M,K](f16) * B[K,Nn](f16, given transposed as BT[Nn,K]) + bias[Nn]
// One wave per 16x16 tile; K stepped by 32 via v_wmma_f32_16x16x32_f16.
// Fragment layouts per CDNA5 ISA 7.12.2:
//   A 16x32 f16 : lane(l&15)=row; halves j=0..7 -> K=k0+hi*8+j ; j=8..15 -> K=k0+16+hi*8+(j-8)
//   B 32x16 f16 : lane(l&15)=col; halves j=0..7 -> K=k0+hi*8+j ; j=8..15 -> K=k0+16+hi*8+(j-8)
//   C/D f32     : lane(l&15)=col; VGPR v -> row v+hi*8
__global__ void __launch_bounds__(32)
gemm16_wmma_kernel(const _Float16* __restrict__ A,   // [M,K]
                   const _Float16* __restrict__ BT,  // [Nn,K]
                   const float* __restrict__ bias,   // [Nn]
                   float* __restrict__ C,            // [M,Nn]
                   int K, int Nn) {
    const int lane = threadIdx.x;
    const int r    = lane & 15;
    const int hi   = lane >> 4;

    const _Float16* arow = A  + (blockIdx.x * 16 + r) * K;
    const _Float16* brow = BT + (blockIdx.y * 16 + r) * K;

    v8f acc = {};
    for (int k0 = 0; k0 < K; k0 += 32) {
        v8h a0 = *(const v8h*)(arow + k0 + hi * 8);        // 16B loads
        v8h a1 = *(const v8h*)(arow + k0 + 16 + hi * 8);
        v8h b0 = *(const v8h*)(brow + k0 + hi * 8);
        v8h b1 = *(const v8h*)(brow + k0 + 16 + hi * 8);
        v16h af, bf;
#pragma unroll
        for (int j = 0; j < 8; ++j) {
            af[j] = a0[j]; af[j + 8] = a1[j];
            bf[j] = b0[j]; bf[j + 8] = b1[j];
        }
        acc = __builtin_amdgcn_wmma_f32_16x16x32_f16(
                  false, af, false, bf, (short)0, acc, false, false);
    }

    const int col = blockIdx.y * 16 + r;
    const float b = bias[col];
    float* crow = C + (blockIdx.x * 16 + hi * 8) * Nn + col;
#pragma unroll
    for (int v = 0; v < 8; ++v) crow[v * Nn] = acc[v] + b;
}

// ------------------------------ edge kernels -------------------------------

// logit[e,h] = att[h,:] . leaky_relu(XL[src,h,:] + XR[dst,h,:], 0.2); segment max into mbuf
template <int C>
__global__ void edge_logit_kernel(const int* __restrict__ src, const int* __restrict__ dst,
                                  const float* __restrict__ XL, const float* __restrict__ XR,
                                  const float* __restrict__ att, float* __restrict__ logit,
                                  unsigned int* __restrict__ mbuf, int nEH) {
    int idx = blockIdx.x * blockDim.x + threadIdx.x;
    if (idx >= nEH) return;
    const int h = idx & (H - 1);
    const int e = idx >> 2;
    const int s = src[e], d = dst[e];
    const float4* xl = (const float4*)(XL + s * (H * C) + h * C);
    const float4* xr = (const float4*)(XR + d * (H * C) + h * C);
    const float4* at = (const float4*)(att + h * C);
    float acc = 0.f;
#pragma unroll
    for (int c = 0; c < C / 4; ++c) {
        float4 a = xl[c], b = xr[c], w = at[c];
        float v;
        v = a.x + b.x; v = v > 0.f ? v : 0.2f * v; acc += w.x * v;
        v = a.y + b.y; v = v > 0.f ? v : 0.2f * v; acc += w.y * v;
        v = a.z + b.z; v = v > 0.f ? v : 0.2f * v; acc += w.z * v;
        v = a.w + b.w; v = v > 0.f ? v : 0.2f * v; acc += w.w * v;
    }
    logit[idx] = acc;
    atomicMax(&mbuf[d * H + h], encode_f32(acc));    // monotone-encoded float max
}

// a = exp(logit - m[dst]); segment sum into zbuf; logit overwritten with a
__global__ void edge_exp_kernel(const int* __restrict__ dst, float* __restrict__ logit,
                                const unsigned int* __restrict__ mbuf,
                                float* __restrict__ zbuf, int nEH) {
    int idx = blockIdx.x * blockDim.x + threadIdx.x;
    if (idx >= nEH) return;
    const int h = idx & (H - 1);
    const int e = idx >> 2;
    const int d = dst[e];
    float m = decode_f32(mbuf[d * H + h]);
    float a = expf(logit[idx] - m);
    logit[idx] = a;
    atomicAdd(&zbuf[d * H + h], a);
}

// out[dst,h,:] += XL[src,h,:] * (a / (z[dst,h] + eps))
template <int C>
__global__ void edge_aggr_kernel(const int* __restrict__ src, const int* __restrict__ dst,
                                 const float* __restrict__ XL, const float* __restrict__ logit,
                                 const float* __restrict__ zbuf, float* __restrict__ outacc,
                                 int nEH) {
    int idx = blockIdx.x * blockDim.x + threadIdx.x;
    if (idx >= nEH) return;
    const int h = idx & (H - 1);
    const int e = idx >> 2;
    const int s = src[e], d = dst[e];
    float w = logit[idx] / (zbuf[d * H + h] + 1e-16f);
    const float* xl = XL + s * (H * C) + h * C;
    float* oa = outacc + d * (H * C) + h * C;
#pragma unroll
    for (int c = 0; c < C; ++c) atomicAdd(&oa[c], xl[c] * w);   // L2-resident atomics
}

// ------------------------------ epilogues ----------------------------------

__global__ void bias_elu_kernel(const float* __restrict__ acc, const float* __restrict__ bo,
                                float* __restrict__ out, int ncols, int n) {
    int i = blockIdx.x * blockDim.x + threadIdx.x;
    if (i >= n) return;
    float v = acc[i] + bo[i % ncols];
    out[i] = v > 0.f ? v : expm1f(v);
}

// log_softmax over 64 columns: one wave (32 lanes) per node, 2 cols per lane
__global__ void __launch_bounds__(256)
logsoftmax64_kernel(const float* __restrict__ acc, const float* __restrict__ bo,
                    float* __restrict__ out, int n) {
    int node = blockIdx.x * (blockDim.x >> 5) + (threadIdx.x >> 5);
    int lane = threadIdx.x & 31;
    if (node >= n) return;
    const float* r = acc + node * 64;
    float v0 = r[lane]      + bo[lane];
    float v1 = r[lane + 32] + bo[lane + 32];
    float mx = fmaxf(v0, v1);
#pragma unroll
    for (int o = 16; o > 0; o >>= 1) mx = fmaxf(mx, __shfl_xor(mx, o, 32));
    float s = expf(v0 - mx) + expf(v1 - mx);
#pragma unroll
    for (int o = 16; o > 0; o >>= 1) s += __shfl_xor(s, o, 32);
    float lse = mx + logf(s);
    out[node * 64 + lane]      = v0 - lse;
    out[node * 64 + lane + 32] = v1 - lse;
}

// ------------------------------ layer driver -------------------------------

template <int C>
static void run_layer(const float* feat, int K,
                      const float* Wl, const float* bl, const float* Wr, const float* br,
                      const float* att, const float* bo,
                      const int* srcA, const int* dstA, int EA,
                      _Float16* xh, _Float16* WlT, _Float16* WrT,
                      float* XL, float* XRacc, float* logit,
                      unsigned int* mbuf, float* zbuf,
                      float* out, bool last, hipStream_t stream) {
    const int HC = H * C;
    const int nf = N_NODES * K;
    cvt_f16_kernel<<<(nf + 255) / 256, 256, 0, stream>>>(feat, xh, nf);
    cvt_w_trans_kernel<<<(K * HC + 255) / 256, 256, 0, stream>>>(Wl, WlT, K, HC);
    cvt_w_trans_kernel<<<(K * HC + 255) / 256, 256, 0, stream>>>(Wr, WrT, K, HC);

    dim3 g(N_NODES / 16, HC / 16);
    gemm16_wmma_kernel<<<g, 32, 0, stream>>>(xh, WlT, bl, XL,    K, HC);
    gemm16_wmma_kernel<<<g, 32, 0, stream>>>(xh, WrT, br, XRacc, K, HC);

    const int nNH = N_NODES * H;
    fill_u32_kernel<<<(nNH + 255) / 256, 256, 0, stream>>>(mbuf, ENC_NEG_INF, nNH);
    fill_u32_kernel<<<(nNH + 255) / 256, 256, 0, stream>>>((unsigned int*)zbuf, 0u, nNH);

    const int nEH = EA * H;
    edge_logit_kernel<C><<<(nEH + 255) / 256, 256, 0, stream>>>(
        srcA, dstA, XL, XRacc, att, logit, mbuf, nEH);

    // XR dead after logit pass -> reuse its buffer as the aggregation accumulator
    const int nNHC = N_NODES * HC;
    fill_u32_kernel<<<(nNHC + 255) / 256, 256, 0, stream>>>((unsigned int*)XRacc, 0u, nNHC);

    edge_exp_kernel<<<(nEH + 255) / 256, 256, 0, stream>>>(dstA, logit, mbuf, zbuf, nEH);
    edge_aggr_kernel<C><<<(nEH + 255) / 256, 256, 0, stream>>>(
        srcA, dstA, XL, logit, zbuf, XRacc, nEH);

    if (!last) {
        bias_elu_kernel<<<(nNHC + 255) / 256, 256, 0, stream>>>(XRacc, bo, out, HC, nNHC);
    } else {
        int blocks = (N_NODES + 7) / 8;   // 8 waves (nodes) per 256-thread block
        logsoftmax64_kernel<<<blocks, 256, 0, stream>>>(XRacc, bo, out, N_NODES);
    }
}

// ------------------------------ entry point --------------------------------

extern "C" void kernel_launch(void* const* d_in, const int* in_sizes, int n_in,
                              void* d_out, int out_size, void* d_ws, size_t ws_size,
                              hipStream_t stream) {
    (void)n_in; (void)out_size; (void)ws_size;

    const float* x  = (const float*)d_in[0];
    const int*   ei = (const int*)d_in[1];
    const int E  = in_sizes[1] / 2;
    const int EA = E + N_NODES;

    const float *Wl0=(const float*)d_in[2],  *bl0=(const float*)d_in[3],
                *Wr0=(const float*)d_in[4],  *br0=(const float*)d_in[5],
                *att0=(const float*)d_in[6], *bo0=(const float*)d_in[7];
    const float *Wl1=(const float*)d_in[8],  *bl1=(const float*)d_in[9],
                *Wr1=(const float*)d_in[10], *br1=(const float*)d_in[11],
                *att1=(const float*)d_in[12],*bo1=(const float*)d_in[13];
    const float *Wl2=(const float*)d_in[14], *bl2=(const float*)d_in[15],
                *Wr2=(const float*)d_in[16], *br2=(const float*)d_in[17],
                *att2=(const float*)d_in[18],*bo2=(const float*)d_in[19];

    // workspace carve-out (256B aligned)
    char* ws = (char*)d_ws;
    size_t off = 0;
    auto carve = [&](size_t bytes) -> char* {
        char* p = ws + off;
        off += (bytes + 255) & ~(size_t)255;
        return p;
    };
    int*          srcA  = (int*)carve((size_t)EA * 4);
    int*          dstA  = (int*)carve((size_t)EA * 4);
    _Float16*     xh    = (_Float16*)carve((size_t)N_NODES * 128 * 2);
    _Float16*     WlT   = (_Float16*)carve((size_t)128 * 128 * 2);
    _Float16*     WrT   = (_Float16*)carve((size_t)128 * 128 * 2);
    float*        XL    = (float*)carve((size_t)N_NODES * 128 * 4);
    float*        XR    = (float*)carve((size_t)N_NODES * 128 * 4);  // reused as accumulator
    float*        logit = (float*)carve((size_t)EA * H * 4);
    unsigned int* mbuf  = (unsigned int*)carve((size_t)N_NODES * H * 4);
    float*        zbuf  = (float*)carve((size_t)N_NODES * H * 4);
    float*        hcur  = (float*)carve((size_t)N_NODES * 128 * 4);

    build_edges_kernel<<<(EA + 255) / 256, 256, 0, stream>>>(ei, srcA, dstA, E, N_NODES);

    // layer 0: 128 -> 4x32, elu
    run_layer<32>(x, 128, Wl0, bl0, Wr0, br0, att0, bo0,
                  srcA, dstA, EA, xh, WlT, WrT, XL, XR, logit, mbuf, zbuf,
                  hcur, false, stream);
    // layer 1: 128 -> 4x32, elu
    run_layer<32>(hcur, 128, Wl1, bl1, Wr1, br1, att1, bo1,
                  srcA, dstA, EA, xh, WlT, WrT, XL, XR, logit, mbuf, zbuf,
                  hcur, false, stream);
    // layer 2: 128 -> 4x16, log_softmax -> d_out
    run_layer<16>(hcur, 128, Wl2, bl2, Wr2, br2, att2, bo2,
                  srcA, dstA, EA, xh, WlT, WrT, XL, XR, logit, mbuf, zbuf,
                  (float*)d_out, true, stream);
}
